// AdaptiveMoELLM_71545565217042
// MI455X (gfx1250) — matrix operand order repaired and checked by reading, not compiled, not run on hardware
//
#include <hip/hip_runtime.h>
#include <hip/hip_bf16.h>
#include <math.h>

// ---- model constants ----
static constexpr int Bb = 2, Ss = 1024, Dd = 512, Hh = 8, DHh = 64;
static constexpr int HIi = 4, DIi = 64, Ff = 2048, Ee = 8, Vv = 32000, Kk = 256;
static constexpr int Tt = Bb * Ss;
#define NEGV (-1e9f)

typedef __attribute__((ext_vector_type(16))) _Float16 v16h;
typedef __attribute__((ext_vector_type(8)))  float    v8f;

union AF { v16h v; uint4 q[2]; };

__device__ __forceinline__ v8f wmma16(v16h a, v16h b, v8f c) {
  return __builtin_amdgcn_wmma_f32_16x16x32_f16(false, a, false, b, (short)0, c,
                                                false, false);
}

// =====================================================================
// f32 -> f16 cast (weights; one pass, then reused by all M-tiles)
// =====================================================================
__global__ __launch_bounds__(256) void cast_k(const float* __restrict__ s,
                                              _Float16* __restrict__ d) {
  size_t i = (size_t)blockIdx.x * 256 + threadIdx.x;
  d[i] = (_Float16)s[i];
}

// =====================================================================
// f16 WMMA GEMM: out[M,N] (+)= epi(A[M,K]h @ W[K,N]h + bias[N])
// block 256 = 8 waves; tile BM=128 BN=128 BK=32; wave = 32x64 (8 wmma/step)
// A staged row-major, W staged transposed -> both fragment loads are b128.
// Dims must be multiples of BM/BN/BK (true for this model).
// EPI: 0=f32 store, 1=gelu->f16, 2=f32 residual add, 3=gated f32 add, 4=f16 store
// =====================================================================
template <int EPI>
__global__ __launch_bounds__(256) void gemm16_k(
    const _Float16* __restrict__ A, const _Float16* __restrict__ W,
    const float* __restrict__ bias, float* __restrict__ outf,
    _Float16* __restrict__ outh, const float* __restrict__ gate,
    int M, int N, int K) {
  __shared__ alignas(16) _Float16 As[128][40];
  __shared__ alignas(16) _Float16 Wt[128][40];  // transposed: Wt[n][k]
  const int tid = threadIdx.x;
  const int wave = tid >> 5, lane = tid & 31;
  const int lr = lane & 15, hi = lane >> 4;
  const int mBase = blockIdx.y * 128;
  const int nBase = blockIdx.x * 128;
  const int wm = (wave & 3) * 32;   // 4 m-slots of 32 rows
  const int wn = (wave >> 2) * 64;  // 2 n-slots of 64 cols
  const v8f vzero = {0.f, 0.f, 0.f, 0.f, 0.f, 0.f, 0.f, 0.f};
  v8f acc[2][4];
#pragma unroll
  for (int mt = 0; mt < 2; ++mt)
#pragma unroll
    for (int nt = 0; nt < 4; ++nt) acc[mt][nt] = vzero;

  for (int k0 = 0; k0 < K; k0 += 32) {
    // A tile 128x32 f16: 512 x uint4 chunks, 2 per thread (b128 in/out)
#pragma unroll
    for (int i = 0; i < 2; ++i) {
      int ch = tid + i * 256;
      int r = ch >> 2, sg = ch & 3;
      *(uint4*)&As[r][sg * 8] =
          *(const uint4*)(A + (size_t)(mBase + r) * K + k0 + sg * 8);
    }
    // W tile 32x128 f16 -> transposed scatter into Wt
#pragma unroll
    for (int i = 0; i < 2; ++i) {
      int ch = tid + i * 256;
      int r = ch >> 4, sg = ch & 15;
      const _Float16* wp = W + (size_t)(k0 + r) * N + nBase + sg * 8;
#pragma unroll
      for (int e = 0; e < 8; ++e) Wt[sg * 8 + e][r] = wp[e];
    }
    if (k0 + 32 < K)
      __builtin_prefetch(A + (size_t)(mBase + wave * 16) * K + k0 + 32, 0, 1);
    __syncthreads();
    AF am[2], bn[4];
#pragma unroll
    for (int mt = 0; mt < 2; ++mt) {
      int m0 = wm + mt * 16 + lr;
      am[mt].q[0] = *(const uint4*)&As[m0][hi * 16];
      am[mt].q[1] = *(const uint4*)&As[m0][hi * 16 + 8];
    }
#pragma unroll
    for (int nt = 0; nt < 4; ++nt) {
      int n0 = wn + nt * 16 + lr;
      bn[nt].q[0] = *(const uint4*)&Wt[n0][hi * 16];
      bn[nt].q[1] = *(const uint4*)&Wt[n0][hi * 16 + 8];
    }
#pragma unroll
    for (int mt = 0; mt < 2; ++mt)
#pragma unroll
      for (int nt = 0; nt < 4; ++nt)
        acc[mt][nt] = wmma16(am[mt].v, bn[nt].v, acc[mt][nt]);
    __syncthreads();
  }
#pragma unroll
  for (int mt = 0; mt < 2; ++mt)
#pragma unroll
    for (int nt = 0; nt < 4; ++nt)
#pragma unroll
      for (int r = 0; r < 8; ++r) {
        int row = mBase + wm + mt * 16 + r + hi * 8;
        int col = nBase + wn + nt * 16 + lr;
        float v = acc[mt][nt][r] + bias[col];
        if (EPI == 0) {
          outf[(size_t)row * N + col] = v;
        } else if (EPI == 1) {  // tanh-approx gelu -> f16
          float c3 = v * v * v;
          float gl = 0.5f * v *
                     (1.f + tanhf(0.7978845608028654f * (v + 0.044715f * c3)));
          outh[(size_t)row * N + col] = (_Float16)gl;
        } else if (EPI == 2) {
          outf[(size_t)row * N + col] += v;
        } else if (EPI == 3) {
          outf[(size_t)row * N + col] += gate[(size_t)row * Ee] * v;
        } else {
          outh[(size_t)row * N + col] = (_Float16)v;
        }
      }
}

// =====================================================================
// Embedding: x[t,d] = tok_emb[ids[t],d] + pos_emb[t%S,d]  (f32 residual)
// =====================================================================
__global__ __launch_bounds__(256) void embed_k(const int* __restrict__ ids,
                                               const float* __restrict__ tok,
                                               const float* __restrict__ pos,
                                               float* __restrict__ x) {
  int i = blockIdx.x * 256 + threadIdx.x;
  int t = i / Dd, d = i - t * Dd;
  int s = t & (Ss - 1);
  x[i] = tok[(size_t)ids[t] * Dd + d] + pos[(size_t)s * Dd + d];
}

// =====================================================================
// LayerNorm (f32 in, f16 out): one block per row
// =====================================================================
__global__ __launch_bounds__(256) void ln_k(const float* __restrict__ x,
                                            const float* __restrict__ g,
                                            const float* __restrict__ b,
                                            _Float16* __restrict__ out) {
  __shared__ float s1[256], s2[256];
  int row = blockIdx.x, tid = threadIdx.x;
  float a0 = x[(size_t)row * Dd + tid];
  float a1 = x[(size_t)row * Dd + 256 + tid];
  s1[tid] = a0 + a1;
  s2[tid] = a0 * a0 + a1 * a1;
  __syncthreads();
  for (int s = 128; s > 0; s >>= 1) {
    if (tid < s) { s1[tid] += s1[tid + s]; s2[tid] += s2[tid + s]; }
    __syncthreads();
  }
  float mu = s1[0] * (1.f / Dd);
  float var = s2[0] * (1.f / Dd) - mu * mu;
  float inv = rsqrtf(var + 1e-5f);
  out[(size_t)row * Dd + tid] = (_Float16)((a0 - mu) * inv * g[tid] + b[tid]);
  out[(size_t)row * Dd + 256 + tid] =
      (_Float16)((a1 - mu) * inv * g[256 + tid] + b[256 + tid]);
}

// =====================================================================
// Final LN variant (f32 out for logits GEMM? not needed) -- f16 used too.
// =====================================================================

// =====================================================================
// Lightning indexer: scores + exact k-th largest threshold (K=256)
// grid (S/16, B), block 256 (8 waves): block owns a 16xS score strip.
// =====================================================================
__global__ __launch_bounds__(256) void indexer_k(
    const _Float16* __restrict__ qi, const _Float16* __restrict__ ki,
    const float* __restrict__ hw, float* __restrict__ scores,
    float* __restrict__ thr) {
  __shared__ alignas(16) float sc[16][1024];
  __shared__ alignas(16) _Float16 qs[16][72];
  __shared__ float red[16][16];
  __shared__ float redt[16];
  const int qt = blockIdx.x, bb = blockIdx.y;
  const int tid = threadIdx.x, wave = tid >> 5, lane = tid & 31;
  const int lr = lane & 15, hi = lane >> 4;

  for (int i = tid; i < 16 * 1024; i += 256) (&sc[0][0])[i] = 0.f;

  for (int h = 0; h < HIi; ++h) {
    __syncthreads();
    for (int i = tid; i < 16 * 64; i += 256) {  // stage Qi tile (f16 copy)
      int r = i >> 6, d = i & 63;
      qs[r][d] = qi[(size_t)(bb * Ss + qt * 16 + r) * (HIi * DIi) + h * DIi + d];
    }
    __syncthreads();
    float wh = hw[h];
    for (int ct = 0; ct < 8; ++ct) {
      int colBase = wave * 128 + ct * 16;
      v8f acc = {0.f, 0.f, 0.f, 0.f, 0.f, 0.f, 0.f, 0.f};
#pragma unroll
      for (int kc = 0; kc < 2; ++kc) {
        int k0 = kc * 32;
        AF a, b;
        a.q[0] = *(const uint4*)&qs[lr][k0 + hi * 16];
        a.q[1] = *(const uint4*)&qs[lr][k0 + hi * 16 + 8];
        size_t kidx = (size_t)(bb * Ss + colBase + lr) * (HIi * DIi) +
                      h * DIi + k0 + hi * 16;
        b.q[0] = *(const uint4*)(ki + kidx);
        b.q[1] = *(const uint4*)(ki + kidx + 8);
        acc = wmma16(a.v, b.v, acc);
      }
#pragma unroll
      for (int r = 0; r < 8; ++r)
        sc[r + hi * 8][colBase + lr] += wh * fmaxf(acc[r], 0.f);
    }
  }
  __syncthreads();

  const int row = tid >> 4, g = tid & 15;
  const int qglob = qt * 16 + row;
  unsigned* su = (unsigned*)&sc[0][0];
  for (int i = 0; i < 64; ++i) {
    int col = g + 16 * i;
    float v = sc[row][col];
    scores[(size_t)(bb * Ss + qglob) * Ss + col] = v;
    float val = v + ((col > qglob) ? NEGV : 0.f);
    unsigned bits = __float_as_uint(val);
    su[row * 1024 + col] = (bits & 0x80000000u) ? ~bits : (bits | 0x80000000u);
  }
  __syncthreads();
  unsigned ans = 0u;
  for (int bit = 31; bit >= 0; --bit) {
    unsigned trial = ans | (1u << bit);
    int cnt = 0;
    for (int i = 0; i < 64; ++i)
      cnt += (su[row * 1024 + g + 16 * i] >= trial) ? 1 : 0;
    red[row][g] = (float)cnt;
    __syncthreads();
    if (g == 0) {
      float tot = 0.f;
      for (int j = 0; j < 16; ++j) tot += red[row][j];
      redt[row] = tot;
    }
    __syncthreads();
    if (redt[row] >= (float)Kk) ans = trial;
    __syncthreads();
  }
  if (g == 0) {
    unsigned bits = (ans & 0x80000000u) ? (ans ^ 0x80000000u) : ~ans;
    thr[bb * Ss + qglob] = __uint_as_float(bits);
  }
}

// =====================================================================
// Sparse-masked attention for one (b, h, 16-query tile), f16 operands.
// QK^T (WMMA) -> masked softmax in LDS -> P(f16 LDS) @ V (WMMA) -> ao f16
// =====================================================================
__global__ __launch_bounds__(256) void attn_k(
    const _Float16* __restrict__ q, const _Float16* __restrict__ k,
    const _Float16* __restrict__ v, const float* __restrict__ scores,
    const float* __restrict__ thr, _Float16* __restrict__ ao) {
  __shared__ alignas(16) float lg[16][1024];
  __shared__ alignas(16) _Float16 ps[16][1024];
  __shared__ alignas(16) _Float16 qs[16][72];
  __shared__ float red[16][16];
  __shared__ float rowmx[16], rowsum[16];
  const int qt = blockIdx.x, hh = blockIdx.y, bb = blockIdx.z;
  const int tid = threadIdx.x, wave = tid >> 5, lane = tid & 31;
  const int lr = lane & 15, hi = lane >> 4;

  for (int i = tid; i < 16 * 64; i += 256) {  // stage Q tile (f16 copy)
    int r = i >> 6, d = i & 63;
    qs[r][d] = q[(size_t)(bb * Ss + qt * 16 + r) * Dd + hh * DHh + d];
  }
  __syncthreads();
  // ---- logits = Q @ K^T ----
  for (int ct = 0; ct < 8; ++ct) {
    int colBase = wave * 128 + ct * 16;
    v8f acc = {0.f, 0.f, 0.f, 0.f, 0.f, 0.f, 0.f, 0.f};
#pragma unroll
    for (int kc = 0; kc < 2; ++kc) {
      int k0 = kc * 32;
      AF a, b;
      a.q[0] = *(const uint4*)&qs[lr][k0 + hi * 16];
      a.q[1] = *(const uint4*)&qs[lr][k0 + hi * 16 + 8];
      size_t kidx =
          (size_t)(bb * Ss + colBase + lr) * Dd + hh * DHh + k0 + hi * 16;
      b.q[0] = *(const uint4*)(k + kidx);
      b.q[1] = *(const uint4*)(k + kidx + 8);
      acc = wmma16(a.v, b.v, acc);
    }
#pragma unroll
    for (int r = 0; r < 8; ++r) lg[r + hi * 8][colBase + lr] = acc[r];
  }
  __syncthreads();
  // ---- masked softmax (causal AND score>=thr) ----
  const int row = tid >> 4, g = tid & 15;
  const int qglob = qt * 16 + row;
  float th = thr[bb * Ss + qglob];
  float pmax = -3.0e38f;
  for (int i = 0; i < 64; ++i) {
    int col = g + 16 * i;
    float sm = scores[(size_t)(bb * Ss + qglob) * Ss + col];
    float maskv = (col > qglob) ? NEGV : ((sm >= th) ? 0.f : NEGV);
    float val = lg[row][col] * 0.125f + maskv;  // 1/sqrt(64)
    lg[row][col] = val;
    pmax = fmaxf(pmax, val);
  }
  red[row][g] = pmax;
  __syncthreads();
  if (g == 0) {
    float m = red[row][0];
    for (int j = 1; j < 16; ++j) m = fmaxf(m, red[row][j]);
    rowmx[row] = m;
  }
  __syncthreads();
  float rm = rowmx[row], psum = 0.f;
  for (int i = 0; i < 64; ++i) {
    int col = g + 16 * i;
    float e = __expf(lg[row][col] - rm);
    lg[row][col] = e;
    psum += e;
  }
  red[row][g] = psum;
  __syncthreads();
  if (g == 0) {
    float s = 0.f;
    for (int j = 0; j < 16; ++j) s += red[row][j];
    rowsum[row] = s;
  }
  __syncthreads();
  float inv = 1.f / rowsum[row];
  for (int i = 0; i < 64; ++i) {  // normalized probs -> f16 tile (once)
    int col = g + 16 * i;
    ps[row][col] = (_Float16)(lg[row][col] * inv);
  }
  __syncthreads();
  // ---- out = P @ V (waves 0..3, each a 16-wide d-slice) ----
  if (wave < 4) {
    int n0 = wave * 16;
    v8f acc = {0.f, 0.f, 0.f, 0.f, 0.f, 0.f, 0.f, 0.f};
    for (int kc = 0; kc < 32; ++kc) {
      int k0 = kc * 32;
      AF a;
      a.q[0] = *(const uint4*)&ps[lr][k0 + hi * 16];
      a.q[1] = *(const uint4*)&ps[lr][k0 + hi * 16 + 8];
      v16h b;
#pragma unroll
      for (int j = 0; j < 16; ++j)
        b[j] = v[(size_t)(bb * Ss + k0 + j + hi * 16) * Dd + hh * DHh + n0 + lr];
      acc = wmma16(a.v, b, acc);
    }
#pragma unroll
    for (int r = 0; r < 8; ++r)
      ao[(size_t)(bb * Ss + qt * 16 + r + hi * 8) * Dd + hh * DHh + n0 + lr] =
          (_Float16)acc[r];
  }
}

// =====================================================================
// Router: softmax over 8 experts, top-2 renormalized gate per token
// =====================================================================
__global__ __launch_bounds__(256) void router_k(const _Float16* __restrict__ m,
                                                const float* __restrict__ rw,
                                                const float* __restrict__ rb,
                                                float* __restrict__ gate) {
  int t = blockIdx.x * 256 + threadIdx.x;
  if (t >= Tt) return;
  float lgt[Ee];
#pragma unroll
  for (int e = 0; e < Ee; ++e) lgt[e] = rb[e];
  for (int d = 0; d < Dd; ++d) {
    float mv = (float)m[(size_t)t * Dd + d];
#pragma unroll
    for (int e = 0; e < Ee; ++e) lgt[e] += mv * rw[d * Ee + e];
  }
  float mx = lgt[0];
#pragma unroll
  for (int e = 1; e < Ee; ++e) mx = fmaxf(mx, lgt[e]);
  float sum = 0.f;
#pragma unroll
  for (int e = 0; e < Ee; ++e) { lgt[e] = __expf(lgt[e] - mx); sum += lgt[e]; }
  float invs = 1.f / sum;
#pragma unroll
  for (int e = 0; e < Ee; ++e) lgt[e] *= invs;
  int i1 = 0;
#pragma unroll
  for (int e = 1; e < Ee; ++e) if (lgt[e] > lgt[i1]) i1 = e;
  int i2 = (i1 == 0) ? 1 : 0;
#pragma unroll
  for (int e = 0; e < Ee; ++e)
    if (e != i1 && lgt[e] > lgt[i2]) i2 = e;
  float gs = 1.f / (lgt[i1] + lgt[i2]);
#pragma unroll
  for (int e = 0; e < Ee; ++e)
    gate[(size_t)t * Ee + e] =
        (e == i1) ? lgt[i1] * gs : ((e == i2) ? lgt[i2] * gs : 0.f);
}

// =====================================================================
static inline char* carve(char*& p, size_t bytes) {
  char* r = p;
  p += (bytes + 255) & ~(size_t)255;
  return r;
}

extern "C" void kernel_launch(void* const* d_in, const int* in_sizes, int n_in,
                              void* d_out, int out_size, void* d_ws,
                              size_t ws_size, hipStream_t stream) {
  (void)in_sizes; (void)n_in; (void)out_size; (void)ws_size;
  const int*   ids      = (const int*)d_in[0];
  const float* tok_emb  = (const float*)d_in[1];
  const float* pos_emb  = (const float*)d_in[2];
  const float* ln1_g    = (const float*)d_in[3];
  const float* ln1_b    = (const float*)d_in[4];
  const float* idx_qw   = (const float*)d_in[5];
  const float* idx_qb   = (const float*)d_in[6];
  const float* idx_kw   = (const float*)d_in[7];
  const float* idx_kb   = (const float*)d_in[8];
  const float* idx_hw   = (const float*)d_in[9];
  const float* wq       = (const float*)d_in[10];
  const float* bq       = (const float*)d_in[11];
  const float* wk       = (const float*)d_in[12];
  const float* bk       = (const float*)d_in[13];
  const float* wv       = (const float*)d_in[14];
  const float* bv       = (const float*)d_in[15];
  const float* wo       = (const float*)d_in[16];
  const float* bo       = (const float*)d_in[17];
  const float* ln2_g    = (const float*)d_in[18];
  const float* ln2_b    = (const float*)d_in[19];
  const float* router_w = (const float*)d_in[20];
  const float* router_b = (const float*)d_in[21];
  const float* e_w1     = (const float*)d_in[22];
  const float* e_b1     = (const float*)d_in[23];
  const float* e_w2     = (const float*)d_in[24];
  const float* e_b2     = (const float*)d_in[25];
  const float* lnf_g    = (const float*)d_in[26];
  const float* lnf_b    = (const float*)d_in[27];
  const float* out_w    = (const float*)d_in[28];
  const float* out_b    = (const float*)d_in[29];
  float* out = (float*)d_out;

  char* p = (char*)d_ws;
  float*     x    = (float*)carve(p, sizeof(float) * Tt * Dd);
  _Float16*  h16  = (_Float16*)carve(p, sizeof(_Float16) * Tt * Dd);
  _Float16*  qi16 = (_Float16*)carve(p, sizeof(_Float16) * Tt * HIi * DIi);
  _Float16*  ki16 = (_Float16*)carve(p, sizeof(_Float16) * Tt * HIi * DIi);
  float*     sc   = (float*)carve(p, sizeof(float) * (size_t)Bb * Ss * Ss);
  float*     th   = (float*)carve(p, sizeof(float) * Tt);
  _Float16*  q16  = (_Float16*)carve(p, sizeof(_Float16) * Tt * Dd);
  _Float16*  k16  = (_Float16*)carve(p, sizeof(_Float16) * Tt * Dd);
  _Float16*  v16  = (_Float16*)carve(p, sizeof(_Float16) * Tt * Dd);
  _Float16*  ao16 = (_Float16*)carve(p, sizeof(_Float16) * Tt * Dd);
  float*     gt   = (float*)carve(p, sizeof(float) * Tt * Ee);
  _Float16*  h116 = (_Float16*)carve(p, sizeof(_Float16) * (size_t)Tt * Ff);
  _Float16*  w16  = (_Float16*)carve(p, sizeof(_Float16) * (size_t)Dd * Vv);

  dim3 blk(256);
  auto castW = [&](const float* src, size_t n) {
    cast_k<<<(unsigned)(n / 256), blk, 0, stream>>>(src, w16);
  };

  embed_k<<<(Tt * Dd) / 256, blk, 0, stream>>>(ids, tok_emb, pos_emb, x);

  for (int l = 0; l < 2; ++l) {
    ln_k<<<Tt, blk, 0, stream>>>(x, ln1_g + l * Dd, ln1_b + l * Dd, h16);
    // indexer projections (f16 out)
    castW(idx_qw + (size_t)l * Dd * HIi * DIi, (size_t)Dd * HIi * DIi);
    gemm16_k<4><<<dim3((HIi * DIi) / 128, Tt / 128), blk, 0, stream>>>(
        h16, w16, idx_qb + l * HIi * DIi, nullptr, qi16, nullptr, Tt,
        HIi * DIi, Dd);
    castW(idx_kw + (size_t)l * Dd * HIi * DIi, (size_t)Dd * HIi * DIi);
    gemm16_k<4><<<dim3((HIi * DIi) / 128, Tt / 128), blk, 0, stream>>>(
        h16, w16, idx_kb + l * HIi * DIi, nullptr, ki16, nullptr, Tt,
        HIi * DIi, Dd);
    indexer_k<<<dim3(Ss / 16, Bb), blk, 0, stream>>>(qi16, ki16,
                                                     idx_hw + l * HIi, sc, th);
    // attention projections (f16 out)
    castW(wq + (size_t)l * Dd * Dd, (size_t)Dd * Dd);
    gemm16_k<4><<<dim3(Dd / 128, Tt / 128), blk, 0, stream>>>(
        h16, w16, bq + l * Dd, nullptr, q16, nullptr, Tt, Dd, Dd);
    castW(wk + (size_t)l * Dd * Dd, (size_t)Dd * Dd);
    gemm16_k<4><<<dim3(Dd / 128, Tt / 128), blk, 0, stream>>>(
        h16, w16, bk + l * Dd, nullptr, k16, nullptr, Tt, Dd, Dd);
    castW(wv + (size_t)l * Dd * Dd, (size_t)Dd * Dd);
    gemm16_k<4><<<dim3(Dd / 128, Tt / 128), blk, 0, stream>>>(
        h16, w16, bv + l * Dd, nullptr, v16, nullptr, Tt, Dd, Dd);
    attn_k<<<dim3(Ss / 16, Hh, Bb), blk, 0, stream>>>(q16, k16, v16, sc, th,
                                                      ao16);
    castW(wo + (size_t)l * Dd * Dd, (size_t)Dd * Dd);
    gemm16_k<2><<<dim3(Dd / 128, Tt / 128), blk, 0, stream>>>(
        ao16, w16, bo + l * Dd, x, nullptr, nullptr, Tt, Dd, Dd);

    // MoE
    ln_k<<<Tt, blk, 0, stream>>>(x, ln2_g + l * Dd, ln2_b + l * Dd, h16);
    router_k<<<Tt / 256, blk, 0, stream>>>(h16, router_w + (size_t)l * Dd * Ee,
                                           router_b + l * Ee, gt);
    for (int e = 0; e < Ee; ++e) {
      castW(e_w1 + (size_t)(l * Ee + e) * Dd * Ff, (size_t)Dd * Ff);
      gemm16_k<1><<<dim3(Ff / 128, Tt / 128), blk, 0, stream>>>(
          h16, w16, e_b1 + (size_t)(l * Ee + e) * Ff, nullptr, h116, nullptr,
          Tt, Ff, Dd);
      castW(e_w2 + (size_t)(l * Ee + e) * Ff * Dd, (size_t)Ff * Dd);
      gemm16_k<3><<<dim3(Dd / 128, Tt / 128), blk, 0, stream>>>(
          h116, w16, e_b2 + (size_t)(l * Ee + e) * Dd, x, nullptr, gt + e, Tt,
          Dd, Ff);
    }
  }
  ln_k<<<Tt, blk, 0, stream>>>(x, lnf_g, lnf_b, h16);
  castW(out_w, (size_t)Dd * Vv);
  gemm16_k<0><<<dim3(Vv / 128, Tt / 128), blk, 0, stream>>>(
      h16, w16, out_b, out, nullptr, nullptr, Tt, Vv, Dd);
}
